// MLP_1709396984333
// MI455X (gfx1250) — matrix operand order repaired and checked by reading, not compiled, not run on hardware
//
#include <hip/hip_runtime.h>
#include <hip/hip_bf16.h>

// ---------------------------------------------------------------------------
// Insight: seq_len==1 softmax is identity (softmax over a single score == 1),
// so each MHA block reduces to  (kv @ Wv.T + bv) @ Wo.T + bo, which fuses to
// a single GEMM with Wf = Wo@Wv, bf = Wo@bv + bo.  Everything becomes GEMMs:
//   8x fuse GEMMs (768^3), 8x cascade GEMMs (4096x768x768, bias/res/bn),
//   MLP 4096x6144x512 + 4096x512x128 + N=1 head.
// All GEMMs run on V_WMMA_F32_16X16X32_BF16 (bf16 in, f32 accumulate).
// Wave tile is 32Mx64N: 2 A-frags x 4 B-frags -> 8 WMMAs per 12 b128 loads.
// ---------------------------------------------------------------------------

#define E 768
#define BATCH 4096
#define BN_INV 0.99999499998749993f /* 1/sqrt(1+1e-5) */

typedef __attribute__((ext_vector_type(16))) __bf16        v16bf;
typedef __attribute__((ext_vector_type(8)))  float         v8f;
typedef __attribute__((ext_vector_type(4)))  unsigned int  u32x4;

union Frag {
  u32x4 q[2];
  v16bf v;
};

__device__ __forceinline__ unsigned short f2bf(float x) {
  unsigned int u = __float_as_uint(x);
  unsigned int r = (u + 0x7FFFu + ((u >> 16) & 1u)) >> 16;  // RNE
  return (unsigned short)r;
}
__device__ __forceinline__ float bf2f(unsigned short h) {
  return __uint_as_float(((unsigned int)h) << 16);
}

// A-matrix (16-bit, 16x32, ISA 7.12.2): lane<16 -> K = k0..k0+7 then k0+16..23,
// lane>=16 -> +8.  Two 16-byte chunks.
__device__ __forceinline__ void load_a(Frag& f, const unsigned short* __restrict__ base,
                                       int ld, int row, int k0, int laneHi) {
  const unsigned short* p = base + (long)row * ld + k0 + (laneHi ? 8 : 0);
  f.q[0] = *(const u32x4*)(p);
  f.q[1] = *(const u32x4*)(p + 16);
}
// B-matrix (16-bit, 32x16): lane n (mod 16) holds column n; lanes 0-15 hold
// K = k0..k0+7 then k0+16..23, lanes 16-31 hold +8 (same as A).
__device__ __forceinline__ void load_b(Frag& f, const unsigned short* __restrict__ base,
                                       int ld, int row, int k0, int laneHi) {
  const unsigned short* p = base + (long)row * ld + k0 + (laneHi ? 8 : 0);
  f.q[0] = *(const u32x4*)(p);
  f.q[1] = *(const u32x4*)(p + 16);
}

#define FLAG_RELU  1
#define FLAG_BN    2
#define FLAG_RES   4
#define FLAG_BIAS  8
#define FLAG_STF32 16
#define FLAG_STBF  32

// C[m,n] = epi( sum_k A[m,k] * W[n,k] ),  A:(M,K) bf16, W:(N,K) bf16 row-major.
// Wave tile 32Mx64N; 4 waves/block -> 128Mx64N per block.
__global__ __launch_bounds__(128) void gemm_bf16_wmma(
    const unsigned short* __restrict__ A, int lda,
    const unsigned short* __restrict__ W, int ldw,
    const float* __restrict__ bias,
    const float* __restrict__ res, int ldres,
    const float* __restrict__ bn_g, const float* __restrict__ bn_b,
    float* __restrict__ Cf, int ldcf,
    unsigned short* __restrict__ Cb, long ldcb, long colOff,
    int M, int N, int K, int flags) {
  const int lane   = threadIdx.x & 31;
  const int wave   = threadIdx.x >> 5;
  const int laneHi = lane >> 4;
  const int laneLo = lane & 15;
  const int m0 = blockIdx.y * 128 + wave * 32;
  const int n0 = blockIdx.x * 64;
  if (m0 >= M) return;  // wave-uniform; grids are exact anyway (EXEC stays full)

  v8f c[2][4];
#pragma unroll
  for (int t = 0; t < 2; ++t)
#pragma unroll
    for (int j = 0; j < 4; ++j) c[t][j] = v8f{};

  const int arow0 = m0 + laneLo;
  for (int k0 = 0; k0 < K; k0 += 32) {
    Frag a[2], b[4];
    load_a(a[0], A, lda, arow0,      k0, laneHi);
    load_a(a[1], A, lda, arow0 + 16, k0, laneHi);
#pragma unroll
    for (int j = 0; j < 4; ++j)
      load_b(b[j], W, ldw, n0 + j * 16 + laneLo, k0, laneHi);
    if (k0 + 32 < K) {  // hint next A lines toward cache (global_prefetch_b8)
      __builtin_prefetch(A + (long)arow0 * lda + k0 + 32, 0, 1);
      __builtin_prefetch(A + (long)(arow0 + 16) * lda + k0 + 32, 0, 1);
    }
#pragma unroll
    for (int t = 0; t < 2; ++t)
#pragma unroll
      for (int j = 0; j < 4; ++j)
        c[t][j] = __builtin_amdgcn_wmma_f32_16x16x32_bf16(
            false, a[t].v, false, b[j].v, (short)0, c[t][j], false, false);
  }

  // Epilogue. C/D layout: VGPR r -> M = mbase + r + 8*laneHi, N = n0c + laneLo.
#pragma unroll
  for (int j = 0; j < 4; ++j) {
    const int n = n0 + j * 16 + laneLo;
    float bv = (flags & FLAG_BIAS) ? bias[n] : 0.0f;
    float gs = 1.0f, gb = 0.0f;
    if (flags & FLAG_BN) { gs = bn_g[n] * BN_INV; gb = bn_b[n]; }
#pragma unroll
    for (int t = 0; t < 2; ++t) {
      const int mbase = m0 + t * 16 + laneHi * 8;
#pragma unroll
      for (int r = 0; r < 8; ++r) {
        const int m = mbase + r;
        float v = c[t][j][r] + bv;
        if (flags & FLAG_RES)  v += res[(long)m * ldres + n];
        if (flags & FLAG_RELU) v = v > 0.0f ? v : 0.0f;
        if (flags & FLAG_BN)   v = v * gs + gb;
        if (flags & FLAG_STF32) Cf[(long)m * ldcf + n] = v;
        if (flags & FLAG_STBF)  Cb[(long)m * ldcb + colOff + n] = f2bf(v);
      }
    }
  }
}

// --- small helper kernels ---------------------------------------------------

__global__ void cvt_f32_bf16_k(const float* __restrict__ src,
                               unsigned short* __restrict__ dst, long n) {
  long i = (long)blockIdx.x * blockDim.x + threadIdx.x;
  long stride = (long)gridDim.x * blockDim.x;
  for (; i < n; i += stride) dst[i] = f2bf(src[i]);
}

// dst[b][k][e] = Win[b][1536+e][k]  (extract Wv and transpose), bf16 out.
__global__ void cvt_wvT_k(const float* __restrict__ Win,
                          unsigned short* __restrict__ dst, int nblk) {
  long total = (long)nblk * E * E;
  long i = (long)blockIdx.x * blockDim.x + threadIdx.x;
  long stride = (long)gridDim.x * blockDim.x;
  for (; i < total; i += stride) {
    int b = (int)(i / (E * E));
    int r = (int)(i % (E * E));
    int k = r / E, e = r % E;
    dst[(long)b * E * E + (long)k * E + e] =
        f2bf(Win[(long)b * 3 * E * E + (long)(1536 + e) * E + k]);
  }
}

// bf[b][n] = sum_e Wo[b][n][e] * bin[b][1536+e] + bo[b][n]
__global__ void fuse_bias_k(const float* __restrict__ Wo,
                            const float* __restrict__ bin,
                            const float* __restrict__ bo,
                            float* __restrict__ bf, int nblk) {
  int idx = blockIdx.x * blockDim.x + threadIdx.x;
  if (idx >= nblk * E) return;
  int b = idx / E, n = idx % E;
  const float* wrow = Wo + (long)b * E * E + (long)n * E;
  const float* bv   = bin + (long)b * 3 * E + 1536;
  float s = 0.0f;
  for (int e = 0; e < E; ++e) s += wrow[e] * bv[e];
  bf[idx] = s + bo[(long)b * E + n];
}

// out[m] = sum_{k<128} H2[m,k] * W3[k] + b3
__global__ void head_k(const unsigned short* __restrict__ H2,
                       const float* __restrict__ W3,
                       const float* __restrict__ b3,
                       float* __restrict__ out, int M) {
  int m = blockIdx.x * blockDim.x + threadIdx.x;
  if (m >= M) return;
  float s = 0.0f;
  for (int k = 0; k < 128; ++k) s += bf2f(H2[(long)m * 128 + k]) * W3[k];
  out[m] = s + b3[0];
}

// ---------------------------------------------------------------------------

extern "C" void kernel_launch(void* const* d_in, const int* in_sizes, int n_in,
                              void* d_out, int out_size, void* d_ws, size_t ws_size,
                              hipStream_t stream) {
  (void)in_sizes; (void)n_in; (void)out_size; (void)ws_size;

  const float* x     = (const float*)d_in[0];
  const float* p_Win = (const float*)d_in[1];
  const float* p_bin = (const float*)d_in[2];
  const float* p_Wo  = (const float*)d_in[3];
  const float* p_bo  = (const float*)d_in[4];
  const float* p_g   = (const float*)d_in[5];
  const float* p_b   = (const float*)d_in[6];
  const float* j_Win = (const float*)d_in[7];
  const float* j_bin = (const float*)d_in[8];
  const float* j_Wo  = (const float*)d_in[9];
  const float* j_bo  = (const float*)d_in[10];
  const float* j_g   = (const float*)d_in[11];
  const float* j_b   = (const float*)d_in[12];
  const float* W1    = (const float*)d_in[13];
  const float* b1    = (const float*)d_in[14];
  const float* g1    = (const float*)d_in[15];
  const float* be1   = (const float*)d_in[16];
  const float* W2    = (const float*)d_in[17];
  const float* b2    = (const float*)d_in[18];
  const float* g2    = (const float*)d_in[19];
  const float* be2   = (const float*)d_in[20];
  const float* W3    = (const float*)d_in[21];
  const float* b3    = (const float*)d_in[22];
  float* out = (float*)d_out;

  // Workspace carve-up (256B aligned)
  char* ws = (char*)d_ws;
  size_t off = 0;
  auto carve = [&](size_t bytes) {
    size_t o = off; off += (bytes + 255) & ~(size_t)255; return o;
  };
  const size_t WBLK = (size_t)E * E;  // 589824 elements
  unsigned short* XB  = (unsigned short*)(ws + carve((size_t)BATCH * 8 * E * 2)); // x bf16
  unsigned short* H   = (unsigned short*)(ws + carve((size_t)BATCH * 8 * E * 2)); // concat c's
  unsigned short* WOB = (unsigned short*)(ws + carve(8 * WBLK * 2));  // Wo bf16
  unsigned short* WVT = (unsigned short*)(ws + carve(8 * WBLK * 2));  // Wv^T bf16
  unsigned short* WF  = (unsigned short*)(ws + carve(8 * WBLK * 2));  // fused W bf16
  unsigned short* W1B = (unsigned short*)(ws + carve((size_t)512 * 8 * E * 2));
  unsigned short* W2B = (unsigned short*)(ws + carve((size_t)128 * 512 * 2));
  float*          BF  = (float*)(ws + carve((size_t)8 * E * 4));      // fused bias
  float*          C0  = (float*)(ws + carve((size_t)BATCH * E * 4));  // residual ping
  float*          C1  = (float*)(ws + carve((size_t)BATCH * E * 4));  // residual pong
  unsigned short* H1  = (unsigned short*)(ws + carve((size_t)BATCH * 512 * 2));
  unsigned short* H2  = (unsigned short*)(ws + carve((size_t)BATCH * 128 * 2));

  // ---- stage 0: conversions -------------------------------------------------
  cvt_f32_bf16_k<<<1024, 256, 0, stream>>>(x, XB, (long)BATCH * 8 * E);
  cvt_f32_bf16_k<<<256, 256, 0, stream>>>(p_Wo, WOB, (long)4 * WBLK);
  cvt_f32_bf16_k<<<256, 256, 0, stream>>>(j_Wo, WOB + 4 * WBLK, (long)4 * WBLK);
  cvt_wvT_k<<<256, 256, 0, stream>>>(p_Win, WVT, 4);
  cvt_wvT_k<<<256, 256, 0, stream>>>(j_Win, WVT + 4 * WBLK, 4);
  cvt_f32_bf16_k<<<256, 256, 0, stream>>>(W1, W1B, (long)512 * 8 * E);
  cvt_f32_bf16_k<<<64, 256, 0, stream>>>(W2, W2B, (long)128 * 512);
  fuse_bias_k<<<(8 * E + 255) / 256, 256, 0, stream>>>(p_Wo, p_bin, p_bo, BF, 4);
  fuse_bias_k<<<(4 * E + 255) / 256, 256, 0, stream>>>(j_Wo, j_bin, j_bo, BF + 4 * E, 4);

  // ---- stage 1: fuse Wf[b] = Wo[b] @ Wv[b]  (C[n,k] = sum_e Wo[n,e]*WvT[k,e])
  for (int b = 0; b < 8; ++b) {
    dim3 grid(E / 64, E / 128), blk(128);
    gemm_bf16_wmma<<<grid, blk, 0, stream>>>(
        WOB + (size_t)b * WBLK, E, WVT + (size_t)b * WBLK, E,
        nullptr, nullptr, 0, nullptr, nullptr, nullptr, 0,
        WF + (size_t)b * WBLK, (long)E, 0L,
        E, E, E, FLAG_STBF);
  }

  // ---- stage 2: cascades (p = blocks 0..3, j = 4..7), residual chain in f32 -
  for (int pfx = 0; pfx < 2; ++pfx) {
    const float* g = (pfx == 0) ? p_g : j_g;
    const float* bb = (pfx == 0) ? p_b : j_b;
    const float* resv = nullptr;
    for (int i = 0; i < 4; ++i) {
      int blkid = pfx * 4 + i;
      float* cout = (i & 1) ? C1 : C0;
      int flags = FLAG_BIAS | FLAG_STF32 | FLAG_STBF;
      if (i > 0) flags |= FLAG_RES | FLAG_BN;
      dim3 grid(E / 64, BATCH / 128), blk(128);
      gemm_bf16_wmma<<<grid, blk, 0, stream>>>(
          XB + (size_t)blkid * E, 8 * E,            // A = x slice, lda = 6144
          WF + (size_t)blkid * WBLK, E,             // W = fused weight
          BF + (size_t)blkid * E,                   // fused bias
          resv, E,                                  // residual = previous c (f32)
          g + (size_t)i * E, bb + (size_t)i * E,    // bn params (used if i>0)
          cout, E,                                  // f32 c_i for next residual
          H, (long)(8 * E), (long)(pfx * 4 * E + i * E),  // bf16 into concat H
          BATCH, E, E, flags);
      resv = cout;
    }
  }

  // ---- stage 3: MLP ---------------------------------------------------------
  {
    dim3 grid(512 / 64, BATCH / 128), blk(128);
    gemm_bf16_wmma<<<grid, blk, 0, stream>>>(
        H, 8 * E, W1B, 8 * E, b1, nullptr, 0, g1, be1,
        nullptr, 0, H1, 512L, 0L,
        BATCH, 512, 8 * E, FLAG_BIAS | FLAG_RELU | FLAG_BN | FLAG_STBF);
  }
  {
    dim3 grid(128 / 64, BATCH / 128), blk(128);
    gemm_bf16_wmma<<<grid, blk, 0, stream>>>(
        H1, 512, W2B, 512, b2, nullptr, 0, g2, be2,
        nullptr, 0, H2, 128L, 0L,
        BATCH, 128, 512, FLAG_BIAS | FLAG_RELU | FLAG_BN | FLAG_STBF);
  }
  head_k<<<(BATCH + 255) / 256, 256, 0, stream>>>(H2, W3, b3, out, BATCH);
}
